// PointMambaBlock_41420664602624
// MI455X (gfx1250) — matrix operand order repaired and checked by reading, not compiled, not run on hardware
//
#include <hip/hip_runtime.h>
#include <cstdint>
#include <cstddef>

// ---------------- problem constants ----------------
#define NROW 53248   // total points (sum of LENGTHS); 53248 = 416*128
#define CDIM 384
#define DIM  768     // DI
#define DST  16      // DS (mamba states)
#define DRK  24      // DR
#define KCV  4       // conv kernel
#define NB   8       // batches

// ---------------- WMMA types ----------------
typedef __attribute__((ext_vector_type(16))) __bf16        v16bf;
typedef __attribute__((ext_vector_type(8)))  float         v8f;
typedef __attribute__((ext_vector_type(8)))  unsigned int  v8u;

union FragBF { v8u u; v16bf b; };

__device__ __forceinline__ unsigned short f2bf(float f) {
  unsigned u = __builtin_bit_cast(unsigned, f);
  u += 0x7FFFu + ((u >> 16) & 1u);           // round-to-nearest-even
  return (unsigned short)(u >> 16);
}
__device__ __forceinline__ float bf2f(unsigned short h) {
  unsigned u = ((unsigned)h) << 16;
  return __builtin_bit_cast(float, u);
}
__device__ __forceinline__ float dsilu(float v) { return v / (1.f + __expf(-v)); }

// ---- CDNA5 async global->LDS copy (ASYNCcnt tracked), 16B per lane ----
__device__ __forceinline__ void async_b128(const unsigned short* base, unsigned byte_off,
                                           void* ldsp) {
  unsigned la = (unsigned)(size_t)ldsp;      // low 32 bits == LDS address
  asm volatile("global_load_async_to_lds_b128 %0, %1, %2"
               :: "v"(la), "v"(byte_off), "s"(base) : "memory");
}
__device__ __forceinline__ void wait_async0() {
  asm volatile("s_wait_asynccnt 0" ::: "memory");
}

// ---------------- bf16 WMMA GEMM:  Y = A @ Wᵀ  (all dims exact multiples) ----------------
// A:  M x Kd bf16 (row-major, stride lda), M % 128 == 0, Kd % 32 == 0
// Wb: Nout x Kd bf16 (row-major, stride ldw), Nout % (NF*32) == 0 (zero-padded)
// NF = 16-wide N-fragments per wave: wave tile = 32 x (NF*16); WG tile = 128 x (NF*32).
// Double-buffered async-to-LDS staging overlapping v_wmma compute.
#define BM 128
#define BK 32
#define LDS_S 48   // LDS row stride in halves (96 B -> 16B-aligned rows)

template<int NF, bool F32OUT, bool BFOUT, bool MAP>
__global__ __launch_bounds__(256)
void gemm_async_wmma(const unsigned short* __restrict__ A, int lda,
                     const unsigned short* __restrict__ Wb, int ldw,
                     float* __restrict__ Yf, unsigned short* __restrict__ Ybf, int ldy,
                     int Kd, const int* __restrict__ row_map)
{
  __shared__ __attribute__((aligned(16))) unsigned short As[2][BM][LDS_S];
  __shared__ __attribute__((aligned(16))) unsigned short Bs[2][NF * 32][LDS_S];
  const int tid  = threadIdx.x;
  const int lane = tid & 31;
  const int wave = tid >> 5;
  const int wm   = (wave & 3) * 32;          // 4 waves tile M
  const int wn   = (wave >> 2) * (NF * 16);  // 2 waves tile N
  const int m0   = blockIdx.x * BM;
  const int n0   = blockIdx.y * (NF * 32);
  const int mfr  = lane & 15;
  const int kb   = lane >> 4;

  // staging coordinates: 16B units; A tile = 512 units (2/thread), B tile = NF*128 units
  const int ar = tid >> 2;            // 0..63
  const int sg = (tid & 3) * 8;       // half offset within row (0,8,16,24)

  auto issue = [&](int k0, int buf) {
    async_b128(A, (unsigned)((((size_t)(m0 + ar)      * lda) + k0 + sg) * 2), &As[buf][ar][sg]);
    async_b128(A, (unsigned)((((size_t)(m0 + 64 + ar) * lda) + k0 + sg) * 2), &As[buf][64 + ar][sg]);
    #pragma unroll
    for (int u = 0; u < NF / 2; ++u) {
      int row = ar + u * 64;
      async_b128(Wb, (unsigned)((((size_t)(n0 + row) * ldw) + k0 + sg) * 2), &Bs[buf][row][sg]);
    }
  };

  v8f acc[2][NF] = {};
  const int ktiles = Kd / BK;
  issue(0, 0);

  for (int kt = 0; kt < ktiles; ++kt) {
    wait_async0();
    __syncthreads();                        // tile kt visible to all waves
    if (kt + 1 < ktiles) issue((kt + 1) * BK, (kt + 1) & 1);  // overlap next DMA

    const int buf = kt & 1;
    FragBF a[2], b[NF];
    #pragma unroll
    for (int i = 0; i < 2; ++i) {
      int r = wm + i * 16 + mfr;            // A frag: lane&15 = M row
      #pragma unroll
      for (int e = 0; e < 8; ++e) {
        int k2 = e * 2;
        int kk = (k2 < 8) ? (k2 + 8 * kb) : (8 + k2 + 8 * kb);
        a[i].u[e] = *(const unsigned*)&As[buf][r][kk];
      }
    }
    #pragma unroll
    for (int j = 0; j < NF; ++j) {
      int c = wn + j * 16 + mfr;            // B frag: lane&15 = N col
      #pragma unroll
      for (int e = 0; e < 8; ++e) {
        int kk = (e * 2 < 8) ? (e * 2 + 8 * kb) : (8 + e * 2 + 8 * kb);
        b[j].u[e] = *(const unsigned*)&Bs[buf][c][kk];
      }
    }
    #pragma unroll
    for (int i = 0; i < 2; ++i)
      #pragma unroll
      for (int j = 0; j < NF; ++j)
        acc[i][j] = __builtin_amdgcn_wmma_f32_16x16x32_bf16(
            false, a[i].b, false, b[j].b, (short)0, acc[i][j], false, false);
  }

  // epilogue: C/D layout row = r + 8*(lane>=16), col = lane&15
  #pragma unroll
  for (int i = 0; i < 2; ++i)
    #pragma unroll
    for (int j = 0; j < NF; ++j) {
      int rbase = m0 + wm + i * 16 + 8 * kb;
      int col   = n0 + wn + j * 16 + mfr;
      #pragma unroll
      for (int r = 0; r < 8; ++r) {
        int row = rbase + r;
        int orow = MAP ? row_map[row] : row;
        float v = acc[i][j][r];
        if (F32OUT) Yf [(size_t)orow * ldy + col] = v;
        if (BFOUT)  Ybf[(size_t)orow * ldy + col] = f2bf(v);
      }
    }
}

// ---------------- elementwise / reduction kernels ----------------
__global__ void zero_kernel(float* p, int n) {
  int g = blockIdx.x * 256 + threadIdx.x;
  if (g < n) p[g] = 0.f;
}

__global__ void f32_to_bf16_kernel(const float* __restrict__ in,
                                   unsigned short* __restrict__ outp, int n) {
  int g = blockIdx.x * 256 + threadIdx.x;
  if (g < n) outp[g] = f2bf(in[g]);
}

// zero-padded f32 -> bf16 weight conversion (rout*cout elements)
__global__ void pad_w_kernel(const float* __restrict__ W, unsigned short* __restrict__ outp,
                             int rin, int cin, int cout, int total) {
  int g = blockIdx.x * 256 + threadIdx.x;
  if (g >= total) return;
  int r = g / cout, c = g - r * cout;
  float v = (r < rin && c < cin) ? W[(size_t)r * cin + c] : 0.f;
  outp[g] = f2bf(v);
}

// per-column sum / sumsq over M rows (one block per column)
__global__ __launch_bounds__(256)
void colstats_kernel(const float* __restrict__ Y, int ldy, int M, float* __restrict__ stats) {
  int c = blockIdx.x;
  float s = 0.f, q = 0.f;
  for (int r = threadIdx.x; r < M; r += 256) {
    float v = Y[(size_t)r * ldy + c];
    s += v; q += v * v;
  }
  __shared__ float sh[512];
  sh[threadIdx.x] = s; sh[256 + threadIdx.x] = q;
  __syncthreads();
  for (int o = 128; o > 0; o >>= 1) {
    if (threadIdx.x < o) {
      sh[threadIdx.x]       += sh[threadIdx.x + o];
      sh[256 + threadIdx.x] += sh[256 + threadIdx.x + o];
    }
    __syncthreads();
  }
  if (threadIdx.x == 0) { stats[2 * c] = sh[0]; stats[2 * c + 1] = sh[256]; }
}

__global__ void bn_finalize_kernel(const float* __restrict__ stats,
                                   const float* __restrict__ g, const float* __restrict__ b,
                                   float invM, float* __restrict__ ab, int Nc) {
  int c = blockIdx.x * 256 + threadIdx.x;
  if (c < Nc) {
    float mean = stats[2 * c] * invM;
    float var  = stats[2 * c + 1] * invM - mean * mean;
    float a    = g[c] * rsqrtf(var + 1e-5f);
    ab[2 * c]     = a;
    ab[2 * c + 1] = b[c] - mean * a;
  }
}

// xs[i,:] = relu(bn(y1[order[i],:]))  (f32 -> bf16)
__global__ void bn_relu_gather_kernel(const float* __restrict__ y1, const float* __restrict__ ab,
                                      const int* __restrict__ order, unsigned short* __restrict__ xs) {
  int gid = blockIdx.x * 256 + threadIdx.x;
  if (gid >= NROW * CDIM) return;
  int i = gid / CDIM, c = gid - i * CDIM;
  int src = order[i];
  float v = ab[2 * c] * y1[(size_t)src * CDIM + c] + ab[2 * c + 1];
  xs[gid] = f2bf(fmaxf(v, 0.f));
}

// causal depthwise conv (K=4) + SiLU ; xm = first DIM cols of xz (stride 2*DIM)
__global__ void conv_silu_kernel(const float* __restrict__ xz, const float* __restrict__ cw,
                                 const float* __restrict__ cb, const int* __restrict__ pos,
                                 unsigned short* __restrict__ xcbf) {
  int gid = blockIdx.x * 256 + threadIdx.x;
  if (gid >= NROW * DIM) return;
  int i = gid / DIM, d = gid - i * DIM;
  int t = pos[i];
  float acc = cb[d];
  #pragma unroll
  for (int k = 0; k < KCV; ++k) {
    int tk = t - (KCV - 1) + k;
    if (tk >= 0)
      acc += xz[(size_t)(i - (KCV - 1 - k)) * (2 * DIM) + d] * cw[d * KCV + k];
  }
  xcbf[gid] = f2bf(dsilu(acc));
}

// dt = softplus(dtpre + b_dt), in place
__global__ void softplus_kernel(float* __restrict__ dt, const float* __restrict__ b_dt) {
  int gid = blockIdx.x * 256 + threadIdx.x;
  if (gid >= NROW * DIM) return;
  int d = gid % DIM;
  float v = dt[gid] + b_dt[d];
  dt[gid] = (v > 20.f) ? v : log1pf(__expf(v));
}

// selective scan: 24 WGs = 8 batches x 3 channel blocks; state resident in VGPRs
__global__ __launch_bounds__(256)
void scan_kernel(const float* __restrict__ dt, const unsigned short* __restrict__ xcbf,
                 const float* __restrict__ dbc, const float* __restrict__ A_log,
                 const int* __restrict__ o, float* __restrict__ ys) {
  int b = blockIdx.x / 3;
  int d = (blockIdx.x % 3) * 256 + threadIdx.x;
  int start = (b == 0) ? 0 : o[b - 1];
  int end   = o[b];
  float An[DST], h[DST];
  #pragma unroll
  for (int s = 0; s < DST; ++s) { An[s] = -__expf(A_log[d * DST + s]); h[s] = 0.f; }
  for (int r = start; r < end; ++r) {
    size_t rb = (size_t)r;
    float dtv = dt[rb * DIM + d];
    float u   = bf2f(xcbf[rb * DIM + d]);
    float Bv[DST], Cv[DST];
    const float4* Bp = (const float4*)(dbc + rb * 64 + DRK);
    const float4* Cp = (const float4*)(dbc + rb * 64 + DRK + DST);
    #pragma unroll
    for (int q = 0; q < 4; ++q) {
      float4 bb = Bp[q], cc = Cp[q];
      Bv[4*q+0]=bb.x; Bv[4*q+1]=bb.y; Bv[4*q+2]=bb.z; Bv[4*q+3]=bb.w;
      Cv[4*q+0]=cc.x; Cv[4*q+1]=cc.y; Cv[4*q+2]=cc.z; Cv[4*q+3]=cc.w;
    }
    float du = dtv * u, y = 0.f;
    #pragma unroll
    for (int s = 0; s < DST; ++s) {
      float e = __expf(dtv * An[s]);
      h[s] = h[s] * e + du * Bv[s];
      y += h[s] * Cv[s];
    }
    ys[rb * DIM + d] = y;
  }
}

// yfin = (ys + xc*Dp) * silu(z)  (bf16 out, in-place over xcbf)
__global__ void gate_kernel(const float* __restrict__ ys, unsigned short* __restrict__ xcbf,
                            const float* __restrict__ xz, const float* __restrict__ Dp) {
  int gid = blockIdx.x * 256 + threadIdx.x;
  if (gid >= NROW * DIM) return;
  int i = gid / DIM, d = gid - i * DIM;
  float yv = ys[gid] + bf2f(xcbf[gid]) * Dp[d];
  float z  = xz[(size_t)i * (2 * DIM) + DIM + d];
  xcbf[gid] = f2bf(yv * dsilu(z));
}

// out = relu(bn(y2) + identity), in-place on d_out
__global__ void bn_add_relu_kernel(float* __restrict__ y, const float* __restrict__ ab,
                                   const float* __restrict__ xid) {
  int gid = blockIdx.x * 256 + threadIdx.x;
  if (gid >= NROW * CDIM) return;
  int c = gid % CDIM;
  float v = ab[2 * c] * y[gid] + ab[2 * c + 1] + xid[gid];
  y[gid] = fmaxf(v, 0.f);
}

// ---------------- workspace layout (bytes) ----------------
constexpr size_t SZ_XBF  = (size_t)NROW * CDIM * 2;        // x in bf16
constexpr size_t SZ_Y1   = (size_t)NROW * CDIM * 4;        // GEMM1 out f32
constexpr size_t SZ_XS   = SZ_XBF;                         // bn+relu+gather bf16
constexpr size_t OFF_XBF = 0;
constexpr size_t OFF_Y1  = OFF_XBF + SZ_XBF;
constexpr size_t OFF_XS  = OFF_Y1 + SZ_Y1;
constexpr size_t OFF_YS  = 0;                              // scan out aliases xbf|y1|xs (dead)
constexpr size_t OFF_XZ  = OFF_XS + SZ_XS;                 // == NROW*768*4 == scan-out size
constexpr size_t OFF_XC  = OFF_XZ  + (size_t)NROW * 2 * DIM * 4;
constexpr size_t OFF_DBC = OFF_XC  + (size_t)NROW * DIM * 2;
constexpr size_t OFF_DBCB= OFF_DBC + (size_t)NROW * 64 * 4;
constexpr size_t OFF_DT  = OFF_DBCB+ (size_t)NROW * 64 * 2;
constexpr size_t OFF_XRB = OFF_DT;                         // xr bf16 aliases dt (dead after scan)
constexpr size_t OFF_ST  = OFF_DT  + (size_t)NROW * DIM * 4;
// padded bf16 weights
constexpr size_t OFF_W1B  = OFF_ST   + 16384;
constexpr size_t OFF_WINB = OFF_W1B  + (size_t)CDIM * CDIM * 2;
constexpr size_t OFF_WXB  = OFF_WINB + (size_t)2 * DIM * CDIM * 2;
constexpr size_t OFF_WDTB = OFF_WXB  + (size_t)64 * DIM * 2;
constexpr size_t OFF_WOB  = OFF_WDTB + (size_t)DIM * 32 * 2;
constexpr size_t OFF_W2B  = OFF_WOB  + (size_t)CDIM * DIM * 2;

extern "C" void kernel_launch(void* const* d_in, const int* in_sizes, int n_in,
                              void* d_out, int out_size, void* d_ws, size_t ws_size,
                              hipStream_t stream) {
  (void)in_sizes; (void)n_in; (void)out_size; (void)ws_size;
  const float* x      = (const float*)d_in[1];
  const int*   o      = (const int*)d_in[2];
  const int*   order  = (const int*)d_in[3];
  const int*   pos    = (const int*)d_in[6];
  const float* W1     = (const float*)d_in[7];
  const float* g1     = (const float*)d_in[8];
  const float* b1     = (const float*)d_in[9];
  const float* W_in   = (const float*)d_in[10];
  const float* conv_w = (const float*)d_in[11];
  const float* conv_b = (const float*)d_in[12];
  const float* W_x    = (const float*)d_in[13];
  const float* W_dt   = (const float*)d_in[14];
  const float* b_dt   = (const float*)d_in[15];
  const float* A_log  = (const float*)d_in[16];
  const float* Dp     = (const float*)d_in[17];
  const float* W_out  = (const float*)d_in[18];
  const float* W2     = (const float*)d_in[19];
  const float* g2     = (const float*)d_in[20];
  const float* b2     = (const float*)d_in[21];
  float* out = (float*)d_out;

  char* w = (char*)d_ws;
  unsigned short* xbf   = (unsigned short*)(w + OFF_XBF);
  float*          y1    = (float*)         (w + OFF_Y1);
  unsigned short* xs    = (unsigned short*)(w + OFF_XS);
  float*          ysb   = (float*)         (w + OFF_YS);
  float*          xz    = (float*)         (w + OFF_XZ);
  unsigned short* xcbf  = (unsigned short*)(w + OFF_XC);
  float*          dbc   = (float*)         (w + OFF_DBC);
  unsigned short* dbcbf = (unsigned short*)(w + OFF_DBCB);
  float*          dtb   = (float*)         (w + OFF_DT);
  unsigned short* xrbf  = (unsigned short*)(w + OFF_XRB);
  float*          stats1= (float*)         (w + OFF_ST);
  float*          ab1   = stats1 + 2 * CDIM;
  float*          stats2= ab1    + 2 * CDIM;
  float*          ab2   = stats2 + 2 * CDIM;
  unsigned short* W1b   = (unsigned short*)(w + OFF_W1B);
  unsigned short* Winb  = (unsigned short*)(w + OFF_WINB);
  unsigned short* Wxb   = (unsigned short*)(w + OFF_WXB);
  unsigned short* Wdtb  = (unsigned short*)(w + OFF_WDTB);
  unsigned short* Wob   = (unsigned short*)(w + OFF_WOB);
  unsigned short* W2b   = (unsigned short*)(w + OFF_W2B);

  const dim3 blk(256);
  const int mblocks = NROW / BM;                          // 416 exactly
  const int ewC  = (NROW * CDIM + 255) / 256;
  const int ewD  = (NROW * DIM  + 255) / 256;
  const float invM = 1.f / (float)NROW;
  auto gw = [](int n) { return dim3((n + 255) / 256); };

  zero_kernel<<<gw(8 * CDIM), blk, 0, stream>>>(stats1, 8 * CDIM);
  f32_to_bf16_kernel<<<dim3(ewC), blk, 0, stream>>>(x, xbf, NROW * CDIM);

  // weight prep: bf16, zero-padded to tile multiples
  pad_w_kernel<<<gw(CDIM * CDIM),    blk, 0, stream>>>(W1,    W1b,  CDIM, CDIM, CDIM, CDIM * CDIM);
  pad_w_kernel<<<gw(2 * DIM * CDIM), blk, 0, stream>>>(W_in,  Winb, 2 * DIM, CDIM, CDIM, 2 * DIM * CDIM);
  pad_w_kernel<<<gw(64 * DIM),       blk, 0, stream>>>(W_x,   Wxb,  DRK + 2 * DST, DIM, DIM, 64 * DIM);
  pad_w_kernel<<<gw(DIM * 32),       blk, 0, stream>>>(W_dt,  Wdtb, DIM, DRK, 32, DIM * 32);
  pad_w_kernel<<<gw(CDIM * DIM),     blk, 0, stream>>>(W_out, Wob,  CDIM, DIM, DIM, CDIM * DIM);
  pad_w_kernel<<<gw(CDIM * CDIM),    blk, 0, stream>>>(W2,    W2b,  CDIM, CDIM, CDIM, CDIM * CDIM);

  // x1 = x @ W1^T   (128x128 WG tiles)
  gemm_async_wmma<4, true, false, false><<<dim3(mblocks, CDIM / 128), blk, 0, stream>>>(
      xbf, CDIM, W1b, CDIM, y1, nullptr, CDIM, CDIM, nullptr);
  colstats_kernel<<<dim3(CDIM), blk, 0, stream>>>(y1, CDIM, NROW, stats1);
  bn_finalize_kernel<<<gw(CDIM), blk, 0, stream>>>(stats1, g1, b1, invM, ab1, CDIM);
  bn_relu_gather_kernel<<<dim3(ewC), blk, 0, stream>>>(y1, ab1, order, xs);

  // xz = xs @ W_in^T  (N x 1536)
  gemm_async_wmma<4, true, false, false><<<dim3(mblocks, (2 * DIM) / 128), blk, 0, stream>>>(
      xs, CDIM, Winb, CDIM, xz, nullptr, 2 * DIM, CDIM, nullptr);

  conv_silu_kernel<<<dim3(ewD), blk, 0, stream>>>(xz, conv_w, conv_b, pos, xcbf);

  // dbc = xc @ W_x^T  (N x 64, rows 56..63 of W zero-padded; f32 + bf16 outputs)
  gemm_async_wmma<2, true, true, false><<<dim3(mblocks, 1), blk, 0, stream>>>(
      xcbf, DIM, Wxb, DIM, dbc, dbcbf, 64, DIM, nullptr);

  // dtpre = dbc[:, :24] @ W_dt^T  (K padded to 32 with zero W columns)
  gemm_async_wmma<4, true, false, false><<<dim3(mblocks, DIM / 128), blk, 0, stream>>>(
      dbcbf, 64, Wdtb, 32, dtb, nullptr, DIM, 32, nullptr);
  softplus_kernel<<<dim3(ewD), blk, 0, stream>>>(dtb, b_dt);

  scan_kernel<<<dim3(NB * 3), blk, 0, stream>>>(dtb, xcbf, dbc, A_log, o, ysb);
  gate_kernel<<<dim3(ewD), blk, 0, stream>>>(ysb, xcbf, xz, Dp);

  // xr[order[i]] = yfin[i] @ W_out^T   (inverse permutation fused into scatter store)
  gemm_async_wmma<4, false, true, true><<<dim3(mblocks, CDIM / 128), blk, 0, stream>>>(
      xcbf, DIM, Wob, DIM, nullptr, xrbf, CDIM, DIM, order);

  // y2 = xr @ W2^T  -> d_out (f32)
  gemm_async_wmma<4, true, false, false><<<dim3(mblocks, CDIM / 128), blk, 0, stream>>>(
      xrbf, CDIM, W2b, CDIM, out, nullptr, CDIM, CDIM, nullptr);
  colstats_kernel<<<dim3(CDIM), blk, 0, stream>>>(out, CDIM, NROW, stats2);
  bn_finalize_kernel<<<gw(CDIM), blk, 0, stream>>>(stats2, g2, b2, invM, ab2, CDIM);
  bn_add_relu_kernel<<<dim3(ewC), blk, 0, stream>>>(out, ab2, x);
}
